// SentimentNet_26525718020160
// MI455X (gfx1250) — matrix operand order repaired and checked by reading, not compiled, not run on hardware
//
#include <hip/hip_runtime.h>
#include <hip/hip_bf16.h>

// ---------------------------------------------------------------------------
// SentimentNet forward for MI455X (gfx1250, wave32, WMMA)
//   S=500, B=64, V=50257, E=512, H=512, L=2
// ---------------------------------------------------------------------------

typedef __bf16 bf16_t;
typedef __attribute__((ext_vector_type(16))) __bf16 v16bf;
typedef __attribute__((ext_vector_type(8)))  __bf16 v8bf;
typedef __attribute__((ext_vector_type(8)))  float  v8f;
typedef __attribute__((ext_vector_type(4)))  int    v4i;

#define S_LEN 500
#define BATCH 64
#define EDIM  512
#define HDIM  512
#define G4H   2048              // 4*H
#define MROWS (S_LEN * BATCH)   // 32000

union vf8u { v8f v; float f[8]; };

// float -> bf16, round-to-nearest-even (no dependence on compiler bf16 cvt)
__device__ __forceinline__ bf16_t f2bf(float f) {
  unsigned u = __float_as_uint(f);
  unsigned r = (u + 0x7FFFu + ((u >> 16) & 1u)) >> 16;
  unsigned short s = (unsigned short)r;
  return *(bf16_t*)&s;
}

__device__ __forceinline__ v8f wmma_bf16(v16bf a, v16bf b, v8f c) {
  return __builtin_amdgcn_wmma_f32_16x16x32_bf16(false, a, false, b, (short)0, c,
                                                 false, false);
}

// A fragment: A is [M,K] row-major bf16, tile at (m0,k0).
// lanes 0-15 hold row m0+lane, K = k0+{0..15}; lanes 16-31: K = k0+{16..31}
__device__ __forceinline__ v16bf load_frag_a(const bf16_t* __restrict__ A, int lda,
                                             int m0, int k0) {
  int lane = threadIdx.x & 31;
  int m  = m0 + (lane & 15);
  int kb = k0 + ((lane & 16) ? 16 : 0);
  const bf16_t* p = A + (size_t)m * lda + kb;
  union { v16bf v; v8bf h[2]; } u;
  u.h[0] = *(const v8bf*)(p);
  u.h[1] = *(const v8bf*)(p + 8);
  return u.v;
}

// B fragment: B supplied transposed, Bt is [N,K] row-major bf16, tile at (n0,k0).
// lane n<16 holds Bt[n0+n][k0..k0+15]; lane>=16 holds Bt[n0+n-16][k0+16..k0+31]
__device__ __forceinline__ v16bf load_frag_b(const bf16_t* __restrict__ Bt, int ldb,
                                             int n0, int k0) {
  int lane = threadIdx.x & 31;
  int n  = n0 + (lane & 15);
  int kb = k0 + ((lane & 16) ? 16 : 0);
  return *(const v16bf*)(Bt + (size_t)n * ldb + kb);
}

// ---------------------------------------------------------------------------
// Kernel 0: weight prep (fp32->bf16, W_word transpose, bias fuse, init state)
// grid 4096 x 256 covers 2048*512 = 1048576 indices
// ---------------------------------------------------------------------------
__global__ __launch_bounds__(256) void prep_kernel(
    const float* __restrict__ W_ih, const float* __restrict__ W_hh,
    const float* __restrict__ W_word, const float* __restrict__ b_ih,
    const float* __restrict__ b_hh,
    bf16_t* __restrict__ wih_b, bf16_t* __restrict__ whh_b,
    bf16_t* __restrict__ wwt_b, float* __restrict__ bc,
    bf16_t* __restrict__ hbuf, unsigned* __restrict__ bar) {
  size_t i = (size_t)blockIdx.x * blockDim.x + threadIdx.x;
  if (i < (size_t)G4H * HDIM) {       // W_ih / W_hh already [N,K] = [4H,H]
    wih_b[i] = f2bf(W_ih[i]);
    whh_b[i] = f2bf(W_hh[i]);
  }
  if (i < (size_t)HDIM * HDIM) {      // transpose W_word: wwt[n][h] = W_word[h][n]
    size_t h = i >> 9, n = i & 511;
    wwt_b[n * HDIM + h] = f2bf(W_word[i]);
  }
  if (i < G4H) bc[i] = b_ih[i] + b_hh[i];
  if (i < 2 * BATCH * HDIM) hbuf[i] = f2bf(0.0f);   // zero both h ping-pong bufs
  if (i == 0) *bar = 0u;
}

// ---------------------------------------------------------------------------
// Kernel 1: embedding gather + bf16 convert.  grid = 32000 blocks x 256
// ---------------------------------------------------------------------------
__global__ __launch_bounds__(256) void embed_kernel(
    const long long* __restrict__ idx, const float* __restrict__ emb,
    bf16_t* __restrict__ out) {
  long long row = idx[blockIdx.x];
  const float* src = emb + (size_t)row * EDIM;
  bf16_t* dst = out + (size_t)blockIdx.x * EDIM;
  int t = threadIdx.x;
  dst[t]       = f2bf(src[t]);
  dst[t + 256] = f2bf(src[t + 256]);
}

// ---------------------------------------------------------------------------
// Kernel 2: xproj = emb @ W_ih^T + (b_ih + b_hh)
//   M=32000, N=2048, K=512.  Block 256 thr = 8 waves, block tile 64x128,
//   wave tile 32x32 (2x2 wmma accumulators).  grid = (16, 500)
// ---------------------------------------------------------------------------
__global__ __launch_bounds__(256) void xproj_gemm(
    const bf16_t* __restrict__ A, const bf16_t* __restrict__ Bt,
    const float* __restrict__ bias, float* __restrict__ C) {
  int wid = threadIdx.x >> 5;
  int wm = wid & 1, wn = wid >> 1;
  int m0 = blockIdx.y * 64 + wm * 32;
  int n0 = blockIdx.x * 128 + wn * 32;
  v8f a00 = {}, a01 = {}, a10 = {}, a11 = {};
  for (int k0 = 0; k0 < EDIM; k0 += 32) {
    v16bf fa0 = load_frag_a(A, EDIM, m0, k0);
    v16bf fa1 = load_frag_a(A, EDIM, m0 + 16, k0);
    v16bf fb0 = load_frag_b(Bt, EDIM, n0, k0);
    v16bf fb1 = load_frag_b(Bt, EDIM, n0 + 16, k0);
    a00 = wmma_bf16(fa0, fb0, a00);
    a01 = wmma_bf16(fa0, fb1, a01);
    a10 = wmma_bf16(fa1, fb0, a10);
    a11 = wmma_bf16(fa1, fb1, a11);
  }
  int lane = threadIdx.x & 31;
  int col0 = n0 + (lane & 15);
  int rb   = (lane & 16) ? 8 : 0;
  vf8u u00, u01, u10, u11;
  u00.v = a00; u01.v = a01; u10.v = a10; u11.v = a11;
  float bs0 = bias[col0], bs1 = bias[col0 + 16];
#pragma unroll
  for (int i = 0; i < 8; ++i) {
    size_t r0 = (size_t)(m0 + rb + i) * G4H;
    size_t r1 = (size_t)(m0 + 16 + rb + i) * G4H;
    C[r0 + col0]      = u00.f[i] + bs0;
    C[r0 + col0 + 16] = u01.f[i] + bs1;
    C[r1 + col0]      = u10.f[i] + bs0;
    C[r1 + col0 + 16] = u11.f[i] + bs1;
  }
}

// ---------------------------------------------------------------------------
// Kernel 3: persistent LSTM.  grid = 16 blocks x 256 (co-resident; device-wide
// barrier per timestep).  Each wave owns one 16(batch) x 16(hidden) slot and
// computes all 4 gate tiles for it -> cell state c lives in VGPRs for the
// entire 500-step sequence.  h ping-pongs via a global double buffer and is
// staged into LDS each step (async global->LDS path when available) so the
// latency-critical A-fragment loads come from LDS.  Next step's xproj slice
// is prefetched (global_prefetch_b8) to overlap HBM latency with the GEMM.
// ---------------------------------------------------------------------------
__global__ __launch_bounds__(256) void lstm_persistent(
    const float* __restrict__ xproj, const bf16_t* __restrict__ whh,
    bf16_t* __restrict__ hbuf, float* __restrict__ states_f,
    bf16_t* __restrict__ states_b, unsigned* __restrict__ bar) {
  extern __shared__ __align__(32) char smem_raw[];
  bf16_t* h_s = (bf16_t*)smem_raw;                        // [64][512] bf16 = 64KB

  const int wid  = threadIdx.x >> 5;
  const int lane = threadIdx.x & 31;
  const int m0   = (wid & 3) * 16;                       // batch tile
  const int hc0  = blockIdx.x * 32 + (wid >> 2) * 16;    // hidden-col tile
  const int col  = hc0 + (lane & 15);
  const int rb   = m0 + ((lane & 16) ? 8 : 0);
  vf8u c_reg;
#pragma unroll
  for (int i = 0; i < 8; ++i) c_reg.f[i] = 0.0f;

  for (int s = 0; s < S_LEN; ++s) {
    const bf16_t* h_in  = hbuf + (size_t)(s & 1) * (BATCH * HDIM);
    bf16_t*       h_out = hbuf + (size_t)((s + 1) & 1) * (BATCH * HDIM);

    // ---- stage h_in (64KB) into LDS: 4096 x 16B chunks, 16 per thread ----
#if __has_builtin(__builtin_amdgcn_global_load_async_to_lds_b128)
#pragma unroll
    for (int j = 0; j < 16; ++j) {
      int idx = (j * 256 + (int)threadIdx.x) * 8;        // 8 bf16 = 16B
      __builtin_amdgcn_global_load_async_to_lds_b128(
          (__attribute__((address_space(1))) v4i*)(uintptr_t)(h_in + idx),
          (__attribute__((address_space(3))) v4i*)(unsigned)(uintptr_t)(h_s + idx),
          0, 0);
    }
#if __has_builtin(__builtin_amdgcn_s_wait_asynccnt)
    __builtin_amdgcn_s_wait_asynccnt(0);
#else
    asm volatile("s_wait_asynccnt 0x0" ::: "memory");
#endif
#else
#pragma unroll
    for (int j = 0; j < 16; ++j) {
      int idx = (j * 256 + (int)threadIdx.x) * 8;
      *(v8bf*)(h_s + idx) = *(const v8bf*)(h_in + idx);
    }
#endif
    __syncthreads();

    // ---- prefetch next step's xproj slice for this thread's slots ----
    if (s + 1 < S_LEN) {
      const float* xn = xproj + (size_t)(s + 1) * BATCH * G4H;
      if ((lane & 15) == 0) {
#pragma unroll
        for (int i = 0; i < 8; ++i) {
#pragma unroll
          for (int g = 0; g < 4; ++g)
            __builtin_prefetch(xn + (size_t)(rb + i) * G4H + g * HDIM + hc0, 0, 1);
        }
      }
    }

    // ---- z = h_in @ W_hh^T for all 4 gates of this wave's slot ----
    v8f zi = {}, zf = {}, zg = {}, zo = {};
    for (int k0 = 0; k0 < HDIM; k0 += 32) {
      v16bf fa  = load_frag_a(h_s, HDIM, m0, k0);        // from LDS
      v16bf bi  = load_frag_b(whh, HDIM, 0 * HDIM + hc0, k0);
      v16bf bf_ = load_frag_b(whh, HDIM, 1 * HDIM + hc0, k0);
      v16bf bg  = load_frag_b(whh, HDIM, 2 * HDIM + hc0, k0);
      v16bf bo  = load_frag_b(whh, HDIM, 3 * HDIM + hc0, k0);
      zi = wmma_bf16(fa, bi, zi);
      zf = wmma_bf16(fa, bf_, zf);
      zg = wmma_bf16(fa, bg, zg);
      zo = wmma_bf16(fa, bo, zo);
    }
    vf8u ui, uf, ug, uo;
    ui.v = zi; uf.v = zf; ug.v = zg; uo.v = zo;
    const float* xp = xproj + (size_t)s * BATCH * G4H;
#pragma unroll
    for (int i = 0; i < 8; ++i) {
      int b = rb + i;
      const float* xr = xp + (size_t)b * G4H;
      float vi = ui.f[i] + xr[col];
      float vf = uf.f[i] + xr[HDIM + col];
      float vg = ug.f[i] + xr[2 * HDIM + col];
      float vo = uo.f[i] + xr[3 * HDIM + col];
      float ig = 1.0f / (1.0f + __expf(-vi));
      float fg = 1.0f / (1.0f + __expf(-vf));
      float og = 1.0f / (1.0f + __expf(-vo));
      float gg = tanhf(vg);
      float cn = fg * c_reg.f[i] + ig * gg;
      c_reg.f[i] = cn;
      float hn = og * tanhf(cn);
      size_t sidx = ((size_t)s * BATCH + b) * HDIM + col;
      states_f[sidx] = hn;
      states_b[sidx] = f2bf(hn);
      h_out[(size_t)b * HDIM + col] = f2bf(hn);
    }

    // ---- device-wide barrier: all 16 blocks publish h_out before next step
    __syncthreads();
#if __has_builtin(__builtin_amdgcn_s_cluster_barrier)
    // Fast hardware path if dispatched as a 16-WG cluster; architecturally a
    // NOP (S_NOP) when ClusterID==0, so safe under a normal dispatch.
    __builtin_amdgcn_s_cluster_barrier();
#endif
    if (threadIdx.x == 0) {
      __threadfence();
      atomicAdd(bar, 1u);
      unsigned target = (unsigned)(s + 1) * gridDim.x;
      while (__hip_atomic_load(bar, __ATOMIC_ACQUIRE, __HIP_MEMORY_SCOPE_AGENT) <
             target) {
        __builtin_amdgcn_s_sleep(1);
      }
    }
    __syncthreads();
    __threadfence();
  }
}

// ---------------------------------------------------------------------------
// Kernel 4: fused attention score:
//   word_attn[m] = sum_n tanh( (states @ W_word)[m,n] + b_word[n] ) * w_proj[n]
//   M=32000, N=512, K=512.  grid = 2000 blocks (16 rows each) x 256.
//   Wave w covers N columns [64w, 64w+64); cross-wave reduce via LDS atomics.
// ---------------------------------------------------------------------------
__global__ __launch_bounds__(256) void attn_kernel(
    const bf16_t* __restrict__ statesb, const bf16_t* __restrict__ wwt,
    const float* __restrict__ b_word, const float* __restrict__ w_proj,
    float* __restrict__ word_attn) {
  __shared__ float ssum[16];
  int wid  = threadIdx.x >> 5;
  int lane = threadIdx.x & 31;
  int m0   = blockIdx.x * 16;
  int nb   = wid * 64;

  v8f acc0 = {}, acc1 = {}, acc2 = {}, acc3 = {};
  for (int k0 = 0; k0 < HDIM; k0 += 32) {
    v16bf fa = load_frag_a(statesb, HDIM, m0, k0);
    v16bf b0 = load_frag_b(wwt, HDIM, nb, k0);
    v16bf b1 = load_frag_b(wwt, HDIM, nb + 16, k0);
    v16bf b2 = load_frag_b(wwt, HDIM, nb + 32, k0);
    v16bf b3 = load_frag_b(wwt, HDIM, nb + 48, k0);
    acc0 = wmma_bf16(fa, b0, acc0);
    acc1 = wmma_bf16(fa, b1, acc1);
    acc2 = wmma_bf16(fa, b2, acc2);
    acc3 = wmma_bf16(fa, b3, acc3);
  }
  vf8u u[4];
  u[0].v = acc0; u[1].v = acc1; u[2].v = acc2; u[3].v = acc3;
  float part[8];
#pragma unroll
  for (int i = 0; i < 8; ++i) part[i] = 0.0f;
#pragma unroll
  for (int j = 0; j < 4; ++j) {
    int cc = nb + j * 16 + (lane & 15);
    float bw = b_word[cc];
    float wp = w_proj[cc];
#pragma unroll
    for (int i = 0; i < 8; ++i) part[i] += tanhf(u[j].f[i] + bw) * wp;
  }
  // reduce over the 16 lanes of each half-wave (rows differ between halves)
#pragma unroll
  for (int off = 8; off; off >>= 1)
#pragma unroll
    for (int i = 0; i < 8; ++i) part[i] += __shfl_xor(part[i], off, 16);

  if (threadIdx.x < 16) ssum[threadIdx.x] = 0.0f;
  __syncthreads();
  if ((lane & 15) == 0) {
    int rbl = (lane & 16) ? 8 : 0;
#pragma unroll
    for (int i = 0; i < 8; ++i) atomicAdd(&ssum[rbl + i], part[i]);
  }
  __syncthreads();
  if (threadIdx.x < 16) word_attn[m0 + threadIdx.x] = ssum[threadIdx.x];
}

// ---------------------------------------------------------------------------
// Kernel 5: softmax over S per batch column.  grid = 64 blocks x 256.
// ---------------------------------------------------------------------------
__global__ __launch_bounds__(256) void softmax_kernel(
    const float* __restrict__ wa, float* __restrict__ attn) {
  __shared__ float red[256];
  int b = blockIdx.x, t = threadIdx.x;
  float mx = -3.0e38f;
  for (int s = t; s < S_LEN; s += 256) mx = fmaxf(mx, wa[s * BATCH + b]);
  red[t] = mx;
  __syncthreads();
  for (int off = 128; off; off >>= 1) {
    if (t < off) red[t] = fmaxf(red[t], red[t + off]);
    __syncthreads();
  }
  mx = red[0];
  __syncthreads();
  float sum = 0.0f;
  for (int s = t; s < S_LEN; s += 256) sum += __expf(wa[s * BATCH + b] - mx);
  red[t] = sum;
  __syncthreads();
  for (int off = 128; off; off >>= 1) {
    if (t < off) red[t] += red[t + off];
    __syncthreads();
  }
  float inv = 1.0f / red[0];
  for (int s = t; s < S_LEN; s += 256)
    attn[s * BATCH + b] = __expf(wa[s * BATCH + b] - mx) * inv;
}

// ---------------------------------------------------------------------------
// Kernel 6: pooled = sum_s attn[s,b]*states[s,b,:]; out = pooled @ dec_W^T + b
// grid = 64 blocks (one per batch) x 256.
// ---------------------------------------------------------------------------
__global__ __launch_bounds__(256) void pool_decode(
    const float* __restrict__ attn, const float* __restrict__ states,
    const float* __restrict__ decW, const float* __restrict__ decb,
    float* __restrict__ out) {
  __shared__ float r0[256], r1[256];
  int b = blockIdx.x, t = threadIdx.x;
  float p0 = 0.0f, p1 = 0.0f;
  for (int s = 0; s < S_LEN; ++s) {
    float w = attn[s * BATCH + b];
    const float* st = states + ((size_t)s * BATCH + b) * HDIM;
    p0 += w * st[t];
    p1 += w * st[t + 256];
  }
  float d0 = p0 * decW[t] + p1 * decW[t + 256];
  float d1 = p0 * decW[HDIM + t] + p1 * decW[HDIM + t + 256];
  r0[t] = d0;
  r1[t] = d1;
  __syncthreads();
  for (int off = 128; off; off >>= 1) {
    if (t < off) { r0[t] += r0[t + off]; r1[t] += r1[t + off]; }
    __syncthreads();
  }
  if (t == 0) {
    out[b * 2 + 0] = r0[0] + decb[0];
    out[b * 2 + 1] = r1[0] + decb[1];
  }
}

// ---------------------------------------------------------------------------
// Workspace layout (bytes, 256-aligned).  Total ~398.4 MB.
// ---------------------------------------------------------------------------
static constexpr size_t OFF_WIH   = 0;                                   // 2048*512*2
static constexpr size_t OFF_WHH   = OFF_WIH   + (size_t)G4H * HDIM * 2;
static constexpr size_t OFF_WWT   = OFF_WHH   + (size_t)G4H * HDIM * 2;  // 512*512*2
static constexpr size_t OFF_BC    = OFF_WWT   + (size_t)HDIM * HDIM * 2; // 2048*4
static constexpr size_t OFF_HBUF  = OFF_BC    + (size_t)G4H * 4;         // 2*64*512*2
static constexpr size_t OFF_BAR   = OFF_HBUF  + (size_t)2 * BATCH * HDIM * 2;
static constexpr size_t OFF_WATT  = OFF_BAR   + 256;                     // 32000*4
static constexpr size_t OFF_ATTN  = OFF_WATT  + (size_t)MROWS * 4;
static constexpr size_t OFF_EMB   = OFF_ATTN  + (size_t)MROWS * 4;       // 32000*512*2
static constexpr size_t OFF_STB   = OFF_EMB   + (size_t)MROWS * EDIM * 2;
static constexpr size_t OFF_STF   = OFF_STB   + (size_t)MROWS * HDIM * 2;
static constexpr size_t OFF_XPROJ = OFF_STF   + (size_t)MROWS * HDIM * 4;

extern "C" void kernel_launch(void* const* d_in, const int* in_sizes, int n_in,
                              void* d_out, int out_size, void* d_ws, size_t ws_size,
                              hipStream_t stream) {
  (void)in_sizes; (void)n_in; (void)out_size; (void)ws_size;
  const long long* inputs    = (const long long*)d_in[0];
  /* d_in[1] = text_lengths: unused by the reference computation */
  const float* embedding     = (const float*)d_in[2];
  const float* W_ih          = (const float*)d_in[3];
  const float* W_hh          = (const float*)d_in[4];
  const float* b_ih          = (const float*)d_in[5];
  const float* b_hh          = (const float*)d_in[6];
  const float* W_word        = (const float*)d_in[7];
  const float* b_word        = (const float*)d_in[8];
  const float* w_proj        = (const float*)d_in[9];
  const float* dec_W         = (const float*)d_in[10];
  const float* dec_b         = (const float*)d_in[11];
  float* out = (float*)d_out;

  char* ws = (char*)d_ws;
  bf16_t*   wih_b  = (bf16_t*)(ws + OFF_WIH);
  bf16_t*   whh_b  = (bf16_t*)(ws + OFF_WHH);
  bf16_t*   wwt_b  = (bf16_t*)(ws + OFF_WWT);
  float*    bc     = (float*)(ws + OFF_BC);
  bf16_t*   hbuf   = (bf16_t*)(ws + OFF_HBUF);
  unsigned* bar    = (unsigned*)(ws + OFF_BAR);
  float*    watt   = (float*)(ws + OFF_WATT);
  float*    attn   = (float*)(ws + OFF_ATTN);
  bf16_t*   emb_b  = (bf16_t*)(ws + OFF_EMB);
  bf16_t*   st_b   = (bf16_t*)(ws + OFF_STB);
  float*    st_f   = (float*)(ws + OFF_STF);
  float*    xproj  = (float*)(ws + OFF_XPROJ);

  prep_kernel<<<4096, 256, 0, stream>>>(W_ih, W_hh, W_word, b_ih, b_hh,
                                        wih_b, whh_b, wwt_b, bc, hbuf, bar);
  embed_kernel<<<MROWS, 256, 0, stream>>>(inputs, embedding, emb_b);
  dim3 gx(16, MROWS / 64);
  xproj_gemm<<<gx, 256, 0, stream>>>(emb_b, wih_b, bc, xproj);
  lstm_persistent<<<16, 256, BATCH * HDIM * sizeof(bf16_t), stream>>>(
      xproj, whh_b, hbuf, st_f, st_b, bar);
  attn_kernel<<<MROWS / 16, 256, 0, stream>>>(st_b, wwt_b, b_word, w_proj, watt);
  softmax_kernel<<<BATCH, 256, 0, stream>>>(watt, attn);
  pool_decode<<<BATCH, 256, 0, stream>>>(attn, st_f, dec_W, dec_b, out);
}
